// Fingerprint_viz_27367531610661
// MI455X (gfx1250) — matrix operand order repaired and checked by reading, not compiled, not run on hardware
//
#include <hip/hip_runtime.h>
#include <hip/hip_bf16.h>

#define BBv   256
#define LLv   128
#define NBv   6
#define FAv   39
#define FBv   10
#define DDv   200
#define DPv   224   // D padded to K-multiple of 32
#define RRv   3
#define TTv   2
#define NEGV  (-9e8f)
#define NBLK  4     // N-tiles per wave (register blocking)

typedef __attribute__((ext_vector_type(16))) __bf16 v16bf;
typedef __attribute__((ext_vector_type(8)))  __bf16 v8bf;
typedef __attribute__((ext_vector_type(8)))  float  v8f;

__device__ __forceinline__ float lrelu_f(float x){ return x > 0.f ? x : 0.01f * x; }
__device__ __forceinline__ float sigm_f(float x){ return 1.f / (1.f + __expf(-x)); }
__device__ __forceinline__ float elu_f(float x){ return x > 0.f ? x : (__expf(x) - 1.f); }

// ---------------------------------------------------------------------------
// Register-blocked bf16 WMMA GEMM: C[M,N] = act( A[M,K] @ Bpack + bias )
// One wave computes a 16 x (16*NBLK) strip: A fragment loaded once per K-step,
// reused by NBLK v_wmma_f32_16x16x32_bf16 issues (4x less A traffic).
// A: bf16 row-major, row stride lda (K zero-padded to Ktiles*32).
// Optional Aidx: per-row gather index (fuses the neighbor gather).
// Bpack: fragment-packed weights: [(kt*Ntiles+nt)*32 + lane]*16 halves.
// ---------------------------------------------------------------------------
__global__ __launch_bounds__(256)
void gemm_bf16_wmma(const __bf16* __restrict__ A, const int* __restrict__ Aidx,
                    const __bf16* __restrict__ Bp, const float* __restrict__ bias,
                    float* __restrict__ Cf, __bf16* __restrict__ Cbf,
                    int M, int N, int lda, int ldc, int ldcbf,
                    int Ntiles, int Ngroups, int Ktiles, int act)
{
    int wid  = (int)((blockIdx.x * blockDim.x + threadIdx.x) >> 5);
    int lane = (int)(threadIdx.x & 31);
    int Mtiles = M >> 4;
    if (wid >= Mtiles * Ngroups) return;         // uniform per wave -> EXEC stays full
    int tm  = wid / Ngroups;
    int tg  = wid - tm * Ngroups;
    int tn0 = tg * NBLK;

    int  arow = (tm << 4) + (lane & 15);
    long asrc = Aidx ? (long)Aidx[arow] : (long)arow;
    const __bf16* Arow = A + asrc * (long)lda;
    int koff = (lane < 16) ? 0 : 8;              // ISA 16-bit A-matrix lane layout

    // Clamp B-load tile indices so the last (partial) group stays in-bounds;
    // duplicate tiles are discarded at store time.
    int tnc[NBLK];
    #pragma unroll
    for (int i = 0; i < NBLK; ++i) {
        int tn = tn0 + i;
        tnc[i] = (tn < Ntiles) ? tn : (Ntiles - 1);
    }

    v8f acc[NBLK];
    #pragma unroll
    for (int i = 0; i < NBLK; ++i) acc[i] = (v8f){0.f,0.f,0.f,0.f,0.f,0.f,0.f,0.f};

    for (int kt = 0; kt < Ktiles; ++kt) {
        int k0 = kt << 5;
        v8bf alo = *(const v8bf*)(Arow + k0 + koff);        // K = koff .. koff+7
        v8bf ahi = *(const v8bf*)(Arow + k0 + koff + 16);   // K = koff+16 .. koff+23
        v16bf av;
        #pragma unroll
        for (int e = 0; e < 8; ++e) { av[e] = alo[e]; av[8 + e] = ahi[e]; }

        #pragma unroll
        for (int i = 0; i < NBLK; ++i) {
            const v8bf* pb = (const v8bf*)(Bp + (((long)kt * Ntiles + tnc[i]) * 32 + lane) * 16);
            v8bf blo = pb[0], bhi = pb[1];
            v16bf bv;
            #pragma unroll
            for (int e = 0; e < 8; ++e) { bv[e] = blo[e]; bv[8 + e] = bhi[e]; }
            acc[i] = __builtin_amdgcn_wmma_f32_16x16x32_bf16(
                         false, av, false, bv, (short)0, acc[i], false, false);
        }
    }

    int rbase = (tm << 4) + ((lane < 16) ? 0 : 8);  // f32 C/D layout
    #pragma unroll
    for (int i = 0; i < NBLK; ++i) {
        int tn = tn0 + i;
        if (tn >= Ntiles) break;
        int col = (tn << 4) + (lane & 15);
        if (col < N) {
            float bs = bias ? bias[col] : 0.f;
            #pragma unroll
            for (int e = 0; e < 8; ++e) {
                float v = acc[i][e] + bs;
                if (act == 1) v = lrelu_f(v);
                long row = rbase + e;
                if (Cf)  Cf [row * ldc   + col] = v;
                if (Cbf) Cbf[row * ldcbf + col] = (__bf16)v;
            }
        }
    }
}

// Pack fp32 weights into WMMA B-fragment order (zero-padded OOB).
// transpose=0: B[k][n] = W[k*ldw+n]; transpose=1: B[k][n] = W[n*ldw+k].
__global__ void pack_weight(const float* __restrict__ W, __bf16* __restrict__ out,
                            int K, int N, int ldw, int Ntiles, int total, int transpose)
{
    int tid = (int)(blockIdx.x * blockDim.x + threadIdx.x);
    if (tid >= total) return;
    int j    = tid & 15;
    int lane = (tid >> 4) & 31;
    int tile = tid >> 9;
    int nt = tile % Ntiles;
    int kt = tile / Ntiles;
    int g = lane >> 4, nl = lane & 15;
    int n = (nt << 4) + nl;
    int k = (kt << 5) + (g << 3) + (j < 8 ? j : j + 8);  // B frag: K = 32*kt + 8*g + j (j<8), 32*kt + 16 + 8*g + (j-8) (j>=8)
    float v = 0.f;
    if (k < K && n < N)
        v = transpose ? W[(long)n * ldw + k] : W[(long)k * ldw + n];
    out[tid] = (__bf16)v;
}

__global__ void zero_pad224(__bf16* __restrict__ buf, int rows)
{
    int tid = (int)(blockIdx.x * blockDim.x + threadIdx.x);
    int padw = DPv - DDv;
    if (tid >= rows * padw) return;
    int r = tid / padw, c = DDv + tid % padw;
    buf[(long)r * DPv + c] = (__bf16)0.f;
}

// Build A for atom_fc: [B*L, 64] bf16, cols >= 39 zero.
__global__ void prep_atomA(const float* __restrict__ atom, __bf16* __restrict__ out)
{
    int tid = (int)(blockIdx.x * blockDim.x + threadIdx.x);
    if (tid >= BBv * LLv * 64) return;
    int row = tid >> 6, c = tid & 63;
    float v = (c < FAv) ? atom[(long)row * FAv + c] : 0.f;
    out[tid] = (__bf16)v;
}

// Build A for neighbor_fc: gather [atom_nb | bond_nb] -> [B*L*NB, 64] bf16.
__global__ void prep_nbA(const float* __restrict__ atom, const float* __restrict__ bond,
                         const int* __restrict__ adeg, const int* __restrict__ bdeg,
                         __bf16* __restrict__ out)
{
    int tid = (int)(blockIdx.x * blockDim.x + threadIdx.x);
    if (tid >= BBv * LLv * NBv * 64) return;
    int m = tid >> 6, c = tid & 63;
    int b = m / (LLv * NBv);
    float v = 0.f;
    if (c < FAv)            v = atom[((long)b * LLv + adeg[m]) * FAv + c];
    else if (c < FAv + FBv) v = bond[((long)b * LLv + bdeg[m]) * FBv + (c - FAv)];
    out[tid] = (__bf16)v;
}

__global__ void prep_idx(const int* __restrict__ adeg, int* __restrict__ idxmap)
{
    int m = (int)(blockIdx.x * blockDim.x + threadIdx.x);
    if (m >= BBv * LLv * NBv) return;
    idxmap[m] = (m / (LLv * NBv)) * LLv + adeg[m];
}

// Neighbor attention: score = lrelu([cur,nb] @ alignW + b) + pad_mask,
// softmax over NB, * attend_mask. One thread per (b,l).
__global__ void att_align(const __bf16* __restrict__ curbf, const __bf16* __restrict__ nbbf,
                          const int* __restrict__ idxmap, const int* __restrict__ adeg,
                          const float* __restrict__ aW, const float* __restrict__ ab,
                          float* __restrict__ wout, int r0)
{
    int bl = (int)(blockIdx.x * blockDim.x + threadIdx.x);
    if (bl >= BBv * LLv) return;
    const __bf16* cr = curbf + (long)bl * DPv;
    float cd = 0.f;
    for (int d = 0; d < DDv; ++d) cd += (float)cr[d] * aW[d];
    float sc[NBv]; float mx = -3.4e38f;
    for (int j = 0; j < NBv; ++j) {
        int m = bl * NBv + j;
        const __bf16* nr = r0 ? (nbbf + (long)m * DPv)
                              : (nbbf + (long)idxmap[m] * DPv);
        float nd = 0.f;
        for (int d = 0; d < DDv; ++d) nd += (float)nr[d] * aW[DDv + d];
        float s = lrelu_f(cd + nd + ab[0]);
        if (adeg[m] == LLv - 1) s += NEGV;
        sc[j] = s; mx = fmaxf(mx, s);
    }
    float sum = 0.f;
    for (int j = 0; j < NBv; ++j) { sc[j] = __expf(sc[j] - mx); sum += sc[j]; }
    float inv = 1.f / sum;
    for (int j = 0; j < NBv; ++j) {
        int m = bl * NBv + j;
        float wv = sc[j] * inv;
        if (adeg[m] == LLv - 1) wv = 0.f;
        wout[m] = wv;
    }
}

// context = elu( sum_j w_j * nb_t_j ), written as bf16 (pad cols pre-zeroed).
__global__ void att_context(const float* __restrict__ w, const float* __restrict__ nbt,
                            __bf16* __restrict__ ctxbf)
{
    int tid = (int)(blockIdx.x * blockDim.x + threadIdx.x);
    if (tid >= BBv * LLv * DDv) return;
    int bl = tid / DDv, d = tid - bl * DDv;
    float s = 0.f;
    for (int j = 0; j < NBv; ++j)
        s += w[bl * NBv + j] * nbt[(long)(bl * NBv + j) * DDv + d];
    ctxbf[(long)bl * DPv + d] = (__bf16)elu_f(s);
}

// GRUCell gate math; gi/gh already contain biases from the GEMMs.
__global__ void gru_gate(const float* __restrict__ gi, const float* __restrict__ gh,
                         const float* __restrict__ hold,
                         float* __restrict__ hf, __bf16* __restrict__ hbf,
                         float* __restrict__ actf, __bf16* __restrict__ actbf, int M)
{
    int tid = (int)(blockIdx.x * blockDim.x + threadIdx.x);
    if (tid >= M * DDv) return;
    int row = tid / DDv, d = tid - row * DDv;
    long o = (long)row * 600;
    float r = sigm_f(gi[o + d]           + gh[o + d]);
    float z = sigm_f(gi[o + DDv + d]     + gh[o + DDv + d]);
    float n = tanhf (gi[o + 2 * DDv + d] + r * gh[o + 2 * DDv + d]);
    float h = (1.f - z) * n + z * hold[(long)row * DDv + d];
    hf [(long)row * DDv + d] = h;
    hbf[(long)row * DPv + d] = (__bf16)h;
    float a = h > 0.f ? h : 0.f;
    actf [(long)row * DDv + d] = a;
    actbf[(long)row * DPv + d] = (__bf16)a;
}

__global__ void mol_init(const float* __restrict__ actf, const float* __restrict__ mask,
                         float* __restrict__ mf, __bf16* __restrict__ mfbf,
                         float* __restrict__ am)
{
    int tid = (int)(blockIdx.x * blockDim.x + threadIdx.x);
    if (tid >= BBv * DDv) return;
    int b = tid / DDv, d = tid - b * DDv;
    float s = 0.f;
    for (int l = 0; l < LLv; ++l)
        s += actf[((long)b * LLv + l) * DDv + d] * mask[b * LLv + l];
    mf[tid] = s;
    mfbf[(long)b * DPv + d] = (__bf16)s;
    am[tid] = s > 0.f ? s : 0.f;
}

// Mol attention softmax over L=128 atoms; one block (128 threads) per molecule.
__global__ void mol_align(const float* __restrict__ am, const float* __restrict__ actf,
                          const float* __restrict__ mask, const float* __restrict__ mW,
                          const float* __restrict__ mb, float* __restrict__ wout)
{
    int b = (int)blockIdx.x, l = (int)threadIdx.x;
    float s = mb[0];
    for (int d = 0; d < DDv; ++d) s += am[b * DDv + d] * mW[d];
    for (int d = 0; d < DDv; ++d) s += actf[((long)b * LLv + l) * DDv + d] * mW[DDv + d];
    s = lrelu_f(s);
    float mk = mask[b * LLv + l];
    if (mk == 0.f) s += NEGV;
    __shared__ float red[LLv];
    red[l] = s; __syncthreads();
    for (int off = LLv / 2; off > 0; off >>= 1) {
        if (l < off) red[l] = fmaxf(red[l], red[l + off]);
        __syncthreads();
    }
    float mx = red[0]; __syncthreads();
    float e = __expf(s - mx);
    red[l] = e; __syncthreads();
    for (int off = LLv / 2; off > 0; off >>= 1) {
        if (l < off) red[l] += red[l + off];
        __syncthreads();
    }
    wout[b * LLv + l] = e / red[0] * mk;
}

__global__ void mol_context(const float* __restrict__ w, const float* __restrict__ at,
                            __bf16* __restrict__ ctxbf)
{
    int tid = (int)(blockIdx.x * blockDim.x + threadIdx.x);
    if (tid >= BBv * DDv) return;
    int b = tid / DDv, d = tid - b * DDv;
    float s = 0.f;
    for (int l = 0; l < LLv; ++l)
        s += w[b * LLv + l] * at[((long)b * LLv + l) * DDv + d];
    ctxbf[(long)b * DPv + d] = (__bf16)elu_f(s);
}

// out[b] = ((cat @ metric_W + metric_b) @ out_W + out_b); cat = [mf, mf + (R-2)]
__global__ void final_out(const float* __restrict__ mf, const float* __restrict__ metW,
                          const float* __restrict__ metb, const float* __restrict__ outW,
                          const float* __restrict__ outb, float* __restrict__ out)
{
    int b = (int)blockIdx.x, d = (int)threadIdx.x;
    const float dd = (float)(RRv - 2);
    __shared__ float red[256];
    float v = 0.f;
    if (d < DDv) {
        float acc = metb[d];
        for (int k = 0; k < DDv; ++k) {
            float mk = mf[b * DDv + k];
            acc += mk * metW[(long)k * DDv + d] + (mk + dd) * metW[(long)(DDv + k) * DDv + d];
        }
        v = acc * outW[d];
    }
    red[d] = v; __syncthreads();
    for (int off = 128; off > 0; off >>= 1) {
        if (d < off) red[d] += red[d + off];
        __syncthreads();
    }
    if (d == 0) out[b] = red[0] + outb[0];
}

// ---------------------------------------------------------------------------
extern "C" void kernel_launch(void* const* d_in, const int* in_sizes, int n_in,
                              void* d_out, int out_size, void* d_ws, size_t ws_size,
                              hipStream_t stream)
{
    (void)in_sizes; (void)n_in; (void)out_size; (void)ws_size;

    const float* atom_list    = (const float*)d_in[0];
    const float* bond_list    = (const float*)d_in[1];
    const float* atom_mask    = (const float*)d_in[2];
    const int*   adeg         = (const int*)  d_in[3];
    const int*   bdeg         = (const int*)  d_in[4];
    const float* atom_fc_W    = (const float*)d_in[5];
    const float* atom_fc_b    = (const float*)d_in[6];
    const float* nb_fc_W      = (const float*)d_in[7];
    const float* nb_fc_b      = (const float*)d_in[8];
    const float* align_W      = (const float*)d_in[9];
    const float* align_b      = (const float*)d_in[10];
    const float* attend_W     = (const float*)d_in[11];
    const float* attend_b     = (const float*)d_in[12];
    const float* gru_wih      = (const float*)d_in[13];
    const float* gru_whh      = (const float*)d_in[14];
    const float* gru_bih      = (const float*)d_in[15];
    const float* gru_bhh      = (const float*)d_in[16];
    const float* mol_align_W  = (const float*)d_in[17];
    const float* mol_align_b  = (const float*)d_in[18];
    const float* mol_attend_W = (const float*)d_in[19];
    const float* mol_attend_b = (const float*)d_in[20];
    const float* mol_gru_wih  = (const float*)d_in[21];
    const float* mol_gru_whh  = (const float*)d_in[22];
    const float* mol_gru_bih  = (const float*)d_in[23];
    const float* mol_gru_bhh  = (const float*)d_in[24];
    const float* metric_W     = (const float*)d_in[25];
    const float* metric_b     = (const float*)d_in[26];
    const float* out_W        = (const float*)d_in[27];
    const float* out_b        = (const float*)d_in[28];

    const int BL = BBv * LLv;          // 32768
    const int M2 = BL * NBv;           // 196608
    const int NT_D = 13, NT_3D = 38, KT_64 = 2, KT_DP = 7;

    char* p = (char*)d_ws;
    auto take = [&](size_t bytes) { char* r = p; p += (bytes + 255) & ~(size_t)255; return r; };

    __bf16* A_atom  = (__bf16*)take((size_t)BL * 64 * 2);
    __bf16* A_nb    = (__bf16*)take((size_t)M2 * 64 * 2);
    float*  af_f    = (float*) take((size_t)BL * DDv * 4);
    __bf16* af_bf   = (__bf16*)take((size_t)BL * DPv * 2);
    __bf16* nbf_bf  = (__bf16*)take((size_t)M2 * DPv * 2);
    int*    idxmap  = (int*)   take((size_t)M2 * 4);
    float*  watt    = (float*) take((size_t)M2 * 4);
    float*  nbt     = (float*) take((size_t)M2 * DDv * 4);
    __bf16* ctx_bf  = (__bf16*)take((size_t)BL * DPv * 2);
    float*  gi      = (float*) take((size_t)BL * 600 * 4);
    float*  gh      = (float*) take((size_t)BL * 600 * 4);
    float*  h_f     = (float*) take((size_t)BL * DDv * 4);
    __bf16* h_bf    = (__bf16*)take((size_t)BL * DPv * 2);
    float*  act_f   = (float*) take((size_t)BL * DDv * 4);
    __bf16* act_bf  = (__bf16*)take((size_t)BL * DPv * 2);
    float*  at_f    = (float*) take((size_t)BL * DDv * 4);
    float*  wmol    = (float*) take((size_t)BL * 4);
    float*  mf_f    = (float*) take((size_t)BBv * DDv * 4);
    __bf16* mf_bf   = (__bf16*)take((size_t)BBv * DPv * 2);
    float*  am_f    = (float*) take((size_t)BBv * DDv * 4);
    __bf16* am_bf   = (__bf16*)take((size_t)BBv * DPv * 2);
    __bf16* mctx_bf = (__bf16*)take((size_t)BBv * DPv * 2);
    float*  mgi     = (float*) take((size_t)BBv * 600 * 4);
    float*  mgh     = (float*) take((size_t)BBv * 600 * 4);

    const size_t pk_fc  = (size_t)KT_64 * NT_D  * 512;
    const size_t pk_att = (size_t)KT_DP * NT_D  * 512;
    const size_t pk_gru = (size_t)KT_DP * NT_3D * 512;
    __bf16* p_atomfc = (__bf16*)take(pk_fc  * 2);
    __bf16* p_nbfc   = (__bf16*)take(pk_fc  * 2);
    __bf16* p_att    = (__bf16*)take(pk_att * 2 * RRv);
    __bf16* p_wih    = (__bf16*)take(pk_gru * 2 * RRv);
    __bf16* p_whh    = (__bf16*)take(pk_gru * 2 * RRv);
    __bf16* p_matt   = (__bf16*)take(pk_att * 2);
    __bf16* p_mwih   = (__bf16*)take(pk_gru * 2);
    __bf16* p_mwhh   = (__bf16*)take(pk_gru * 2);

    auto blocks = [](long n) { return dim3((unsigned)((n + 255) / 256)); };

    // --- pack weights into WMMA B-fragment order ---
    pack_weight<<<blocks((long)pk_fc), 256, 0, stream>>>(atom_fc_W, p_atomfc, FAv, DDv, DDv, NT_D, (int)pk_fc, 0);
    pack_weight<<<blocks((long)pk_fc), 256, 0, stream>>>(nb_fc_W,   p_nbfc, FAv + FBv, DDv, DDv, NT_D, (int)pk_fc, 0);
    for (int r = 0; r < RRv; ++r) {
        pack_weight<<<blocks((long)pk_att), 256, 0, stream>>>(attend_W + (long)r * DDv * DDv, p_att + r * pk_att, DDv, DDv, DDv, NT_D, (int)pk_att, 0);
        pack_weight<<<blocks((long)pk_gru), 256, 0, stream>>>(gru_wih + (long)r * 600 * DDv, p_wih + r * pk_gru, DDv, 600, DDv, NT_3D, (int)pk_gru, 1);
        pack_weight<<<blocks((long)pk_gru), 256, 0, stream>>>(gru_whh + (long)r * 600 * DDv, p_whh + r * pk_gru, DDv, 600, DDv, NT_3D, (int)pk_gru, 1);
    }
    pack_weight<<<blocks((long)pk_att), 256, 0, stream>>>(mol_attend_W, p_matt, DDv, DDv, DDv, NT_D, (int)pk_att, 0);
    pack_weight<<<blocks((long)pk_gru), 256, 0, stream>>>(mol_gru_wih, p_mwih, DDv, 600, DDv, NT_3D, (int)pk_gru, 1);
    pack_weight<<<blocks((long)pk_gru), 256, 0, stream>>>(mol_gru_whh, p_mwhh, DDv, 600, DDv, NT_3D, (int)pk_gru, 1);

    // --- zero K-padding columns of bf16 activation buffers ---
    zero_pad224<<<blocks((long)BL * 24), 256, 0, stream>>>(af_bf,  BL);
    zero_pad224<<<blocks((long)M2 * 24), 256, 0, stream>>>(nbf_bf, M2);
    zero_pad224<<<blocks((long)BL * 24), 256, 0, stream>>>(ctx_bf, BL);
    zero_pad224<<<blocks((long)BL * 24), 256, 0, stream>>>(h_bf,   BL);
    zero_pad224<<<blocks((long)BL * 24), 256, 0, stream>>>(act_bf, BL);
    zero_pad224<<<blocks((long)BBv * 24), 256, 0, stream>>>(mf_bf,   BBv);
    zero_pad224<<<blocks((long)BBv * 24), 256, 0, stream>>>(am_bf,   BBv);
    zero_pad224<<<blocks((long)BBv * 24), 256, 0, stream>>>(mctx_bf, BBv);

    // --- prep A matrices and neighbor gather indices ---
    prep_atomA<<<blocks((long)BL * 64), 256, 0, stream>>>(atom_list, A_atom);
    prep_nbA  <<<blocks((long)M2 * 64), 256, 0, stream>>>(atom_list, bond_list, adeg, bdeg, A_nb);
    prep_idx  <<<blocks((long)M2), 256, 0, stream>>>(adeg, idxmap);

    auto gemm = [&](const __bf16* A, const int* Aidx, const __bf16* Bp, const float* bias,
                    float* Cf, __bf16* Cbf, int M, int N, int lda, int ldc, int ldcbf,
                    int Ntiles, int Ktiles, int act) {
        int Ngroups = (Ntiles + NBLK - 1) / NBLK;
        long waves = (long)(M >> 4) * Ngroups;
        gemm_bf16_wmma<<<blocks(waves * 32), 256, 0, stream>>>(
            A, Aidx, Bp, bias, Cf, Cbf, M, N, lda, ldc, ldcbf, Ntiles, Ngroups, Ktiles, act);
    };

    // atom_fc: lrelu(atom_list @ W + b) -> atom_feature (f32 + bf16)
    gemm(A_atom, nullptr, p_atomfc, atom_fc_b, af_f, af_bf, BL, DDv, 64, DDv, DPv, NT_D, KT_64, 1);
    // neighbor_fc: lrelu([atom_nb|bond_nb] @ W + b) -> nb_feat (bf16 only)
    gemm(A_nb, nullptr, p_nbfc, nb_fc_b, nullptr, nbf_bf, M2, DDv, 64, DDv, DPv, NT_D, KT_64, 1);

    // --- message passing radii ---
    for (int r = 0; r < RRv; ++r) {
        const __bf16* cur   = (r == 0) ? af_bf  : act_bf;
        const __bf16* nbsrc = (r == 0) ? nbf_bf : act_bf;
        att_align<<<blocks((long)BL), 256, 0, stream>>>(
            cur, nbsrc, idxmap, adeg, align_W + (long)r * 2 * DDv, align_b + r, watt, (r == 0) ? 1 : 0);
        // nb_t = nb_feat @ attend_W[r] + attend_b[r]   (gather fused for r>0)
        gemm(nbsrc, (r == 0) ? nullptr : idxmap, p_att + r * pk_att, attend_b + (long)r * DDv,
             nbt, nullptr, M2, DDv, DPv, DDv, 0, NT_D, KT_DP, 0);
        att_context<<<blocks((long)BL * DDv), 256, 0, stream>>>(watt, nbt, ctx_bf);
        // GRU: gi = ctx @ wih.T + bih ; gh = h @ whh.T + bhh
        gemm(ctx_bf, nullptr, p_wih + r * pk_gru, gru_bih + (long)r * 600, gi, nullptr, BL, 600, DPv, 600, 0, NT_3D, KT_DP, 0);
        gemm((r == 0) ? af_bf : h_bf, nullptr, p_whh + r * pk_gru, gru_bhh + (long)r * 600, gh, nullptr, BL, 600, DPv, 600, 0, NT_3D, KT_DP, 0);
        gru_gate<<<blocks((long)BL * DDv), 256, 0, stream>>>(
            gi, gh, (r == 0) ? af_f : h_f, h_f, h_bf, act_f, act_bf, BL);
    }

    // --- molecule readout ---
    mol_init<<<blocks((long)BBv * DDv), 256, 0, stream>>>(act_f, atom_mask, mf_f, mf_bf, am_f);
    // at = activated @ mol_attend_W + b (loop-invariant over T -> hoisted)
    gemm(act_bf, nullptr, p_matt, mol_attend_b, at_f, nullptr, BL, DDv, DPv, DDv, 0, NT_D, KT_DP, 0);
    for (int t = 0; t < TTv; ++t) {
        mol_align<<<dim3(BBv), dim3(LLv), 0, stream>>>(am_f, act_f, atom_mask, mol_align_W, mol_align_b, wmol);
        mol_context<<<blocks((long)BBv * DDv), 256, 0, stream>>>(wmol, at_f, mctx_bf);
        gemm(mctx_bf, nullptr, p_mwih, mol_gru_bih, mgi, nullptr, BBv, 600, DPv, 600, 0, NT_3D, KT_DP, 0);
        gemm(mf_bf,   nullptr, p_mwhh, mol_gru_bhh, mgh, nullptr, BBv, 600, DPv, 600, 0, NT_3D, KT_DP, 0);
        gru_gate<<<blocks((long)BBv * DDv), 256, 0, stream>>>(mgi, mgh, mf_f, mf_f, mf_bf, am_f, am_bf, BBv);
    }

    final_out<<<dim3(BBv), dim3(256), 0, stream>>>(mf_f, metric_W, metric_b, out_W, out_b, (float*)d_out);
}
